// ComplexGraphAttentionLayer_32530082299955
// MI455X (gfx1250) — matrix operand order repaired and checked by reading, not compiled, not run on hardware
//
#include <hip/hip_runtime.h>
#include <hip/hip_bf16.h>
#include <cstdint>

// ===========================================================================
// ComplexGraphAttentionLayer forward for MI455X (gfx1250, wave32, WMMA).
//
// Roofline: ~232 GFLOP total, ~250 MB unavoidable HBM traffic -> balanced at
// 23.3 TB/s only if the S x S score tensor never hits HBM.  Attention is
// computed flash-style (online softmax).  All GEMMs run through
// v_wmma_f32_16x16x32_bf16 (f32 accumulate); LN/residual math stays in f32.
// GEMM waves are register-blocked 32x32 (2x2 WMMA tiles) so every A/B
// fragment feeds four WMMAs: 16 WMMAs per 8 fragment loads per K-step.
//
// Assumed d_in order (setup_inputs dict order, params flattened in insertion
// order): 0:x 1:graph_mask 2-5:q{Wr,Wi,br,bi} 6-9:k 10-13:v 14-17:o
// 18-21:ffn1 22-25:ffn2 26:bias_weights 27:ln1_g 28:ln1_b 29:ln2_g 30:ln2_b
// ===========================================================================

typedef __attribute__((ext_vector_type(16))) __bf16 v16bf;
typedef __attribute__((ext_vector_type(8)))  __bf16 v8bf;
typedef __attribute__((ext_vector_type(8)))  float  v8f;

#define TOK   2048   // B*S
#define BATCH 2
#define SEQ   1024
#define DM    1024
#define NH    16
#define HD    64
#define FFD   4096

__device__ __forceinline__ v8f wmma_bf16(v16bf a, v16bf b, v8f c) {
  // (neg_a, A, neg_b, B, c_mod, C, reuse_a, reuse_b)
  return __builtin_amdgcn_wmma_f32_16x16x32_bf16(false, a, false, b, (short)0, c,
                                                 false, false);
}

// A fragment (16x32 bf16, MxK): lane<16 holds row=lane,
// K=[k0,k0+16); lane>=16 holds K=[k0+16,k0+32)
__device__ __forceinline__ v16bf frag_a(const __bf16* base, int ld, int lane, int k0) {
  const __bf16* p = base + (size_t)(lane & 15) * ld + (k0 + ((lane >> 4) << 4));
  return *(const v16bf*)p;
}

// B fragment (32x16 bf16, KxN) sourced from a row-major [N,K] matrix
// (K contiguous): lane<16 -> col=lane, K=[k0,k0+16); lane>=16 -> K=[k0+16,k0+32)
__device__ __forceinline__ v16bf frag_b(const __bf16* nmajor, int ld, int lane, int k0) {
  const __bf16* p = nmajor + (size_t)(lane & 15) * ld + (k0 + ((lane >> 4) << 4));
  return *(const v16bf*)p;
}

// ---------------------------------------------------------------------------
// f32 -> bf16 conversion
// ---------------------------------------------------------------------------
__global__ __launch_bounds__(256) void cvt_f32_bf16(const float* __restrict__ s,
                                                    __bf16* __restrict__ d, size_t n) {
  size_t i = (size_t)blockIdx.x * 256 + threadIdx.x;
  if (i < n) d[i] = (__bf16)s[i];
}

// ---------------------------------------------------------------------------
// Complex linear: yr = xr*Wr^T - xi*Wi^T + br ; yi = xr*Wi^T + xi*Wr^T + bi
// Block: 8 waves (4x2) -> 128x64 tile.  Each wave: 32x32 output (2x2 WMMA
// tiles) with four f32 accumulators each (rr, ii, ri, ir): 16 WMMAs per
// 32-wide K step against 8 fragment loads (2x operand reuse in registers).
// ---------------------------------------------------------------------------
template <int DIN, int DOUT, bool RELU, bool F32OUT>
__global__ __launch_bounds__(256) void cgemm_kernel(
    const __bf16* __restrict__ X,   // [TOK, 2*DIN]  (real | imag)
    const __bf16* __restrict__ Wr,  // [DOUT, DIN]
    const __bf16* __restrict__ Wi,  // [DOUT, DIN]
    const float* __restrict__ br, const float* __restrict__ bi,
    __bf16* __restrict__ Yb,        // [TOK, 2*DOUT] (!F32OUT)
    float* __restrict__ Yf)         // [TOK, 2*DOUT] (F32OUT)
{
  const int lane = threadIdx.x & 31;
  const int wave = threadIdx.x >> 5;
  const int m0 = blockIdx.y * 128 + (wave >> 1) * 32;  // 32 token rows / wave
  const int n0 = blockIdx.x * 64 + (wave & 1) * 32;    // 32 out cols / wave

  const __bf16* xr[2] = { X + (size_t)m0 * (2 * DIN),
                          X + (size_t)(m0 + 16) * (2 * DIN) };
  const __bf16* xi[2] = { xr[0] + DIN, xr[1] + DIN };
  const __bf16* wr[2] = { Wr + (size_t)n0 * DIN, Wr + (size_t)(n0 + 16) * DIN };
  const __bf16* wi[2] = { Wi + (size_t)n0 * DIN, Wi + (size_t)(n0 + 16) * DIN };

  v8f rr[2][2] = {}, ii[2][2] = {}, ri[2][2] = {}, ir[2][2] = {};

  for (int k0 = 0; k0 < DIN; k0 += 32) {
    v16bf fxr[2], fxi[2], fwr[2], fwi[2];
#pragma unroll
    for (int t = 0; t < 2; ++t) {
      fxr[t] = frag_a(xr[t], 2 * DIN, lane, k0);
      fxi[t] = frag_a(xi[t], 2 * DIN, lane, k0);
      fwr[t] = frag_b(wr[t], DIN, lane, k0);
      fwi[t] = frag_b(wi[t], DIN, lane, k0);
    }
#pragma unroll
    for (int mi = 0; mi < 2; ++mi)
#pragma unroll
      for (int ni = 0; ni < 2; ++ni) {
        rr[mi][ni] = wmma_bf16(fxr[mi], fwr[ni], rr[mi][ni]);
        ii[mi][ni] = wmma_bf16(fxi[mi], fwi[ni], ii[mi][ni]);
        ri[mi][ni] = wmma_bf16(fxr[mi], fwi[ni], ri[mi][ni]);
        ir[mi][ni] = wmma_bf16(fxi[mi], fwr[ni], ir[mi][ni]);
      }
    if (k0 + 128 < DIN) {  // global_prefetch_b8 hint for a future K tile
      __builtin_prefetch(wr[0] + (size_t)(lane & 15) * DIN + k0 + 128, 0, 0);
      __builtin_prefetch(xr[0] + (size_t)(lane & 15) * 2 * DIN + k0 + 128, 0, 0);
    }
  }

  // C/D layout: VGPR v -> row +v+(lane>=16)*8, col +(lane&15)
  const int hi8 = (lane >> 4) << 3;
#pragma unroll
  for (int mi = 0; mi < 2; ++mi) {
#pragma unroll
    for (int ni = 0; ni < 2; ++ni) {
      const int ocol = n0 + ni * 16 + (lane & 15);
      const int orow0 = m0 + mi * 16 + hi8;
      const float brv = br[ocol], biv = bi[ocol];
#pragma unroll
      for (int v = 0; v < 8; ++v) {
        float yr = rr[mi][ni][v] - ii[mi][ni][v] + brv;
        float yi = ri[mi][ni][v] + ir[mi][ni][v] + biv;
        if (RELU) { yr = fmaxf(yr, 0.f); yi = fmaxf(yi, 0.f); }
        const size_t row = (size_t)(orow0 + v);
        if (F32OUT) {
          Yf[row * (2 * DOUT) + ocol] = yr;
          Yf[row * (2 * DOUT) + DOUT + ocol] = yi;
        } else {
          Yb[row * (2 * DOUT) + ocol] = (__bf16)yr;
          Yb[row * (2 * DOUT) + DOUT + ocol] = (__bf16)yi;
        }
      }
    }
  }
}

// ---------------------------------------------------------------------------
// Flash-style complex attention.  Grid (S/128, NH, B), 8 waves/block, each
// wave owns a 16-query tile.  KV tiles of 32 staged in LDS (V transposed so
// the PV B-operand is K-contiguous).  scores = (Re^2 + Im^2)/sqrt(64) + bias.
// ---------------------------------------------------------------------------
__global__ __launch_bounds__(256) void attn_kernel(
    const __bf16* __restrict__ Q, const __bf16* __restrict__ K,
    const __bf16* __restrict__ V,
    const int* __restrict__ gm,   // [B,S,S]
    const float* __restrict__ bw, // [NH,3]
    __bf16* __restrict__ O)       // [B*S, 2*DM]
{
  const int lane = threadIdx.x & 31;
  const int wave = threadIdx.x >> 5;
  const int h = blockIdx.y;
  const int b = blockIdx.z;
  const int q0 = blockIdx.x * 128 + wave * 16;
  const int ld = 2 * DM;
  const size_t bbase = (size_t)b * SEQ * ld;
  const int hi8 = (lane >> 4) << 3;

  __shared__ alignas(32) __bf16 kt_r[32 * 64];      // [k][d]
  __shared__ alignas(32) __bf16 kt_i[32 * 64];
  __shared__ alignas(32) __bf16 vt_r[64 * 32];      // [d][k] (transposed)
  __shared__ alignas(32) __bf16 vt_i[64 * 32];
  __shared__ alignas(32) __bf16 pbuf[8][16 * 32];   // per-wave P tile

  const __bf16* qr_b = Q + bbase + (size_t)q0 * ld + h * HD;
  const __bf16* qi_b = qr_b + DM;
  const v16bf qr0 = frag_a(qr_b, ld, lane, 0);
  const v16bf qr1 = frag_a(qr_b, ld, lane, 32);
  const v16bf qi0 = frag_a(qi_b, ld, lane, 0);
  const v16bf qi1 = frag_a(qi_b, ld, lane, 32);

  const float bw_local = bw[h * 3 + 0];
  const float bw_hier  = bw[h * 3 + 1];

  v8f o_r[4] = {}, o_i[4] = {};
  float mrun[8], lrun[8];
#pragma unroll
  for (int v = 0; v < 8; ++v) { mrun[v] = -1e30f; lrun[v] = 0.f; }

  const int trow = threadIdx.x >> 3;        // 0..31 (key row)
  const int tseg = (threadIdx.x & 7) << 3;  // 0..56 (dim segment)

  for (int k0g = 0; k0g < SEQ; k0g += 32) {
    // -------- cooperative KV tile load (V transposed on LDS store) --------
    {
      const __bf16* ksrc = K + bbase + (size_t)(k0g + trow) * ld + h * HD + tseg;
      *(v8bf*)&kt_r[trow * 64 + tseg] = *(const v8bf*)ksrc;
      *(v8bf*)&kt_i[trow * 64 + tseg] = *(const v8bf*)(ksrc + DM);
      const __bf16* vsrc = V + bbase + (size_t)(k0g + trow) * ld + h * HD + tseg;
      v8bf vr = *(const v8bf*)vsrc;
      v8bf vi = *(const v8bf*)(vsrc + DM);
#pragma unroll
      for (int j = 0; j < 8; ++j) {
        vt_r[(tseg + j) * 32 + trow] = vr[j];
        vt_i[(tseg + j) * 32 + trow] = vi[j];
      }
    }
    __syncthreads();

    // -------- scores for this wave's 16 q x 32 k (two 16x16 halves) --------
    float s[2][8];
#pragma unroll
    for (int half = 0; half < 2; ++half) {
      const __bf16* ktr = &kt_r[half * 16 * 64];
      const __bf16* kti = &kt_i[half * 16 * 64];
      v16bf kr0 = frag_b(ktr, 64, lane, 0);
      v16bf kr1 = frag_b(ktr, 64, lane, 32);
      v16bf ki0 = frag_b(kti, 64, lane, 0);
      v16bf ki1 = frag_b(kti, 64, lane, 32);
      v8f z = {};
      v8f re = wmma_bf16(qr0, kr0, z);            // Re = qr.kr + qi.ki
      re = wmma_bf16(qr1, kr1, re);
      re = wmma_bf16(qi0, ki0, re);
      re = wmma_bf16(qi1, ki1, re);
      v8f i1 = wmma_bf16(qr0, ki0, z);            // Im = qr.ki - qi.kr
      i1 = wmma_bf16(qr1, ki1, i1);
      v8f i2 = wmma_bf16(qi0, kr0, z);
      i2 = wmma_bf16(qi1, kr1, i2);
      const int kcol = k0g + half * 16 + (lane & 15);
#pragma unroll
      for (int v = 0; v < 8; ++v) {
        const int qrow = q0 + v + hi8;
        const float im = i1[v] - i2[v];
        float sc = (re[v] * re[v] + im * im) * 0.125f;  // 1/sqrt(64)
        int dd = qrow - kcol; dd = dd < 0 ? -dd : dd;
        if (dd <= 16) sc += bw_local;
        if (gm[((size_t)b * SEQ + qrow) * SEQ + kcol] > 1) sc += bw_hier;
        s[half][v] = sc;
      }
    }

    // -------- online softmax; stage P (16x32 bf16) in wave-private LDS ----
    __bf16* pw = pbuf[wave];
    float alpha[8];
#pragma unroll
    for (int v = 0; v < 8; ++v) {
      float mx = fmaxf(s[0][v], s[1][v]);
#pragma unroll
      for (int msk = 1; msk <= 8; msk <<= 1) mx = fmaxf(mx, __shfl_xor(mx, msk, 32));
      const float mnew = fmaxf(mrun[v], mx);
      alpha[v] = __expf(mrun[v] - mnew);
      const float p0 = __expf(s[0][v] - mnew);
      const float p1 = __expf(s[1][v] - mnew);
      float rs = p0 + p1;
#pragma unroll
      for (int msk = 1; msk <= 8; msk <<= 1) rs += __shfl_xor(rs, msk, 32);
      lrun[v] = lrun[v] * alpha[v] + rs;
      mrun[v] = mnew;
      pw[(v + hi8) * 32 + (lane & 15)] = (__bf16)p0;
      pw[(v + hi8) * 32 + 16 + (lane & 15)] = (__bf16)p1;
    }
#pragma unroll
    for (int t = 0; t < 4; ++t)
#pragma unroll
      for (int v = 0; v < 8; ++v) { o_r[t][v] *= alpha[v]; o_i[t][v] *= alpha[v]; }

    // Same-wave LDS store -> cross-lane LDS load: drain DScnt.
    asm volatile("s_wait_dscnt 0" ::: "memory");

    const v16bf pf = frag_a(pw, 32, lane, 0);     // P as A operand, K=32
#pragma unroll
    for (int t = 0; t < 4; ++t) {
      v16bf vfr = frag_b(&vt_r[t * 16 * 32], 32, lane, 0);
      v16bf vfi = frag_b(&vt_i[t * 16 * 32], 32, lane, 0);
      o_r[t] = wmma_bf16(pf, vfr, o_r[t]);
      o_i[t] = wmma_bf16(pf, vfi, o_i[t]);
    }
    __syncthreads();   // protect kt/vt before next tile's cooperative load
  }

  // -------- normalize and store (head-major packing matches reference) ----
#pragma unroll
  for (int v = 0; v < 8; ++v) {
    const float inv = 1.0f / lrun[v];
    const size_t row = (size_t)(q0 + v + hi8);
#pragma unroll
    for (int t = 0; t < 4; ++t) {
      const int d = t * 16 + (lane & 15);
      O[bbase + row * ld + h * HD + d]      = (__bf16)(o_r[t][v] * inv);
      O[bbase + row * ld + DM + h * HD + d] = (__bf16)(o_i[t][v] * inv);
    }
  }
}

// ---------------------------------------------------------------------------
// out = LayerNorm(A + Bv) * g + be ; optional bf16 mirror for next GEMM input
// ---------------------------------------------------------------------------
__global__ __launch_bounds__(256) void resid_ln_kernel(
    const float* __restrict__ A, const float* __restrict__ Bv,
    const float* __restrict__ g, const float* __restrict__ be,
    float* __restrict__ outF, __bf16* __restrict__ outB)
{
  const int t = blockIdx.x;
  const int tid = threadIdx.x;
  const size_t base = (size_t)t * (2 * DM);
  float vals[8];
  float s = 0.f, ss = 0.f;
#pragma unroll
  for (int j = 0; j < 8; ++j) {
    const int i = tid + j * 256;
    const float v = A[base + i] + Bv[base + i];
    vals[j] = v; s += v; ss += v * v;
  }
#pragma unroll
  for (int msk = 1; msk < 32; msk <<= 1) {
    s += __shfl_xor(s, msk, 32); ss += __shfl_xor(ss, msk, 32);
  }
  __shared__ float wsum[8], wsq[8];
  if ((tid & 31) == 0) { wsum[tid >> 5] = s; wsq[tid >> 5] = ss; }
  __syncthreads();
  float S = 0.f, SS = 0.f;
#pragma unroll
  for (int w = 0; w < 8; ++w) { S += wsum[w]; SS += wsq[w]; }
  const float mu = S * (1.f / (2 * DM));
  const float var = SS * (1.f / (2 * DM)) - mu * mu;
  const float rstd = rsqrtf(var + 1e-5f);
#pragma unroll
  for (int j = 0; j < 8; ++j) {
    const int i = tid + j * 256;
    const float o = (vals[j] - mu) * rstd * g[i] + be[i];
    outF[base + i] = o;
    if (outB) outB[base + i] = (__bf16)o;
  }
}

// ---------------------------------------------------------------------------
extern "C" void kernel_launch(void* const* d_in, const int* in_sizes, int n_in,
                              void* d_out, int out_size, void* d_ws, size_t ws_size,
                              hipStream_t stream) {
  (void)in_sizes; (void)n_in; (void)out_size; (void)ws_size;
  const float* xF  = (const float*)d_in[0];
  const int*   gm  = (const int*)d_in[1];
  const float* qWr = (const float*)d_in[2],  *qWi = (const float*)d_in[3];
  const float* qbr = (const float*)d_in[4],  *qbi = (const float*)d_in[5];
  const float* kWr = (const float*)d_in[6],  *kWi = (const float*)d_in[7];
  const float* kbr = (const float*)d_in[8],  *kbi = (const float*)d_in[9];
  const float* vWr = (const float*)d_in[10], *vWi = (const float*)d_in[11];
  const float* vbr = (const float*)d_in[12], *vbi = (const float*)d_in[13];
  const float* oWr = (const float*)d_in[14], *oWi = (const float*)d_in[15];
  const float* obr = (const float*)d_in[16], *obi = (const float*)d_in[17];
  const float* f1Wr = (const float*)d_in[18], *f1Wi = (const float*)d_in[19];
  const float* f1br = (const float*)d_in[20], *f1bi = (const float*)d_in[21];
  const float* f2Wr = (const float*)d_in[22], *f2Wi = (const float*)d_in[23];
  const float* f2br = (const float*)d_in[24], *f2bi = (const float*)d_in[25];
  const float* bw   = (const float*)d_in[26];
  const float* ln1g = (const float*)d_in[27], *ln1b = (const float*)d_in[28];
  const float* ln2g = (const float*)d_in[29], *ln2b = (const float*)d_in[30];

  // ---- workspace carve (~128 MB, 256B aligned) ----
  char* cur = (char*)d_ws;
  auto carve = [&](size_t bytes) -> char* {
    char* p = cur; cur += (bytes + 255) & ~(size_t)255; return p;
  };
  __bf16* wqr = (__bf16*)carve((size_t)DM * DM * 2);
  __bf16* wqi = (__bf16*)carve((size_t)DM * DM * 2);
  __bf16* wkr = (__bf16*)carve((size_t)DM * DM * 2);
  __bf16* wki = (__bf16*)carve((size_t)DM * DM * 2);
  __bf16* wvr = (__bf16*)carve((size_t)DM * DM * 2);
  __bf16* wvi = (__bf16*)carve((size_t)DM * DM * 2);
  __bf16* wor = (__bf16*)carve((size_t)DM * DM * 2);
  __bf16* woi = (__bf16*)carve((size_t)DM * DM * 2);
  __bf16* w1r = (__bf16*)carve((size_t)FFD * DM * 2);
  __bf16* w1i = (__bf16*)carve((size_t)FFD * DM * 2);
  __bf16* w2r = (__bf16*)carve((size_t)DM * FFD * 2);
  __bf16* w2i = (__bf16*)carve((size_t)DM * FFD * 2);
  __bf16* xb  = (__bf16*)carve((size_t)TOK * 2 * DM * 2);
  char*   qkvo_base = carve((size_t)TOK * 2 * DM * 2 * 4);  // qb|kb|vb|ob
  __bf16* qb = (__bf16*)qkvo_base;
  __bf16* kb = qb + (size_t)TOK * 2 * DM;
  __bf16* vb = kb + (size_t)TOK * 2 * DM;
  __bf16* ob = vb + (size_t)TOK * 2 * DM;
  __bf16* hb = (__bf16*)qkvo_base;                    // aliases qkvo (dead by FFN1)
  float*  of  = (float*)carve((size_t)TOK * 2 * DM * 4);
  float*  f2f = of;                                   // aliases of (dead by FFN2)
  float*  x1f = (float*)carve((size_t)TOK * 2 * DM * 4);
  __bf16* x1b = (__bf16*)carve((size_t)TOK * 2 * DM * 2);

  auto cvt = [&](const float* s, __bf16* d, size_t n) {
    cvt_f32_bf16<<<dim3((unsigned)((n + 255) / 256)), 256, 0, stream>>>(s, d, n);
  };
  cvt(xF, xb, (size_t)TOK * 2 * DM);
  cvt(qWr, wqr, (size_t)DM * DM);   cvt(qWi, wqi, (size_t)DM * DM);
  cvt(kWr, wkr, (size_t)DM * DM);   cvt(kWi, wki, (size_t)DM * DM);
  cvt(vWr, wvr, (size_t)DM * DM);   cvt(vWi, wvi, (size_t)DM * DM);
  cvt(oWr, wor, (size_t)DM * DM);   cvt(oWi, woi, (size_t)DM * DM);
  cvt(f1Wr, w1r, (size_t)FFD * DM); cvt(f1Wi, w1i, (size_t)FFD * DM);
  cvt(f2Wr, w2r, (size_t)DM * FFD); cvt(f2Wi, w2i, (size_t)DM * FFD);

  const dim3 blk(256);
  const dim3 g_dd(DM / 64, TOK / 128);   // 1024-out GEMMs: 16 x 16 blocks
  const dim3 g_f1(FFD / 64, TOK / 128);  // ffn1: 64 x 16 blocks

  // Q/K/V projections (bf16 out)
  cgemm_kernel<DM, DM, false, false><<<g_dd, blk, 0, stream>>>(xb, wqr, wqi, qbr, qbi, qb, nullptr);
  cgemm_kernel<DM, DM, false, false><<<g_dd, blk, 0, stream>>>(xb, wkr, wki, kbr, kbi, kb, nullptr);
  cgemm_kernel<DM, DM, false, false><<<g_dd, blk, 0, stream>>>(xb, wvr, wvi, vbr, vbi, vb, nullptr);

  // Flash attention
  attn_kernel<<<dim3(SEQ / 128, NH, BATCH), blk, 0, stream>>>(qb, kb, vb, gm, bw, ob);

  // Output projection (f32 out), residual + LN1
  cgemm_kernel<DM, DM, false, true><<<g_dd, blk, 0, stream>>>(ob, wor, woi, obr, obi, nullptr, of);
  resid_ln_kernel<<<dim3(TOK), blk, 0, stream>>>(xF, of, ln1g, ln1b, x1f, x1b);

  // FFN1 (CReLU fused, bf16 out) -> FFN2 (f32 out) -> residual + LN2
  cgemm_kernel<DM, FFD, true, false><<<g_f1, blk, 0, stream>>>(x1b, w1r, w1i, f1br, f1bi, hb, nullptr);
  cgemm_kernel<FFD, DM, false, true><<<g_dd, blk, 0, stream>>>(hb, w2r, w2i, f2br, f2bi, nullptr, f2f);
  resid_ln_kernel<<<dim3(TOK), blk, 0, stream>>>(x1f, f2f, ln2g, ln2b, (float*)d_out, nullptr);
}
